// GatedAttn_73229192396862
// MI455X (gfx1250) — matrix operand order repaired and checked by reading, not compiled, not run on hardware
//
#include <hip/hip_runtime.h>
#include <hip/hip_bf16.h>

// MI455X / gfx1250, wave32. Strategy:
//  1) Precompute WuqT[b][h][m] = sum_k Wq[h,k]*uq[b,m,k]  (WMMA bf16 GEMM, bf16 out)
//     and    UpWp[b][n][h] = sum_k up[b,n,k]*Wp[h,k]      (WMMA bf16 GEMM, f32 out)
//  2) Pre-transpose all in-loop matvec weights to k-major bf16 (WvT, WgT, wihT, whhT)
//     and uq to uqT[b][i][m] bf16 (L2-resident streams, 64MB total).
//  3) Persistent scan: 64 blocks (one per independent batch) x 1024 threads,
//     1024 sequential steps, all reductions via wave32 shfl + LDS.

typedef __attribute__((ext_vector_type(16))) __bf16 v16bf;
typedef __attribute__((ext_vector_type(8)))  float  v8f;

#define DEV static __device__ __forceinline__

DEV float bf2f(__bf16 v) { return (float)v; }

DEV float fast_tanh(float x) {
#if __has_builtin(__builtin_amdgcn_tanhf)
  return __builtin_amdgcn_tanhf(x);           // v_tanh_f32 on gfx1250
#else
  float e = __expf(2.0f * x);                  // v_exp_f32 path
  return 1.0f - 2.0f / (e + 1.0f);
#endif
}

DEV float fast_sigmoid(float x) { return 1.0f / (1.0f + __expf(-x)); }

DEV v16bf pack16(float4 a, float4 b, float4 c, float4 d) {
  v16bf v;
  v[0]=(__bf16)a.x;  v[1]=(__bf16)a.y;  v[2]=(__bf16)a.z;  v[3]=(__bf16)a.w;
  v[4]=(__bf16)b.x;  v[5]=(__bf16)b.y;  v[6]=(__bf16)b.z;  v[7]=(__bf16)b.w;
  v[8]=(__bf16)c.x;  v[9]=(__bf16)c.y;  v[10]=(__bf16)c.z; v[11]=(__bf16)c.w;
  v[12]=(__bf16)d.x; v[13]=(__bf16)d.y; v[14]=(__bf16)d.z; v[15]=(__bf16)d.w;
  return v;
}

// ---------------------------------------------------------------------------
// f32 (bat,R,C) -> bf16 (bat,C,R) transpose/convert
// ---------------------------------------------------------------------------
__global__ void k_transpose_f32_bf16(const float* __restrict__ in,
                                     __bf16* __restrict__ out,
                                     int batches, int R, int C) {
  long rc = (long)R * C;
  long total = (long)batches * rc;
  for (long idx = (long)blockIdx.x * blockDim.x + threadIdx.x; idx < total;
       idx += (long)gridDim.x * blockDim.x) {
    int b = (int)(idx / rc);
    long rem = idx - (long)b * rc;
    int r = (int)(rem / C);
    int c = (int)(rem - (long)r * C);
    out[(long)b * rc + (long)c * R + r] = (__bf16)in[idx];
  }
}

// ---------------------------------------------------------------------------
// C[M x N] = A[M x K] * B[N x K]^T   (A,B f32 row-major, converted to bf16
// in-register; WMMA f32_16x16x32_bf16; one 16x16 C tile per wave)
// A layout (16x32 bf16): lane row = l&15, K = 8*hi + {0..7} and + {16..23}
// B layout (32x16 bf16): lane col = l&15, K = 8*hi + {0..7} and + {16..23}
// C layout: VGPR r -> row = r + 8*hi, col = l&15
// ---------------------------------------------------------------------------
template <bool BF16OUT>
__global__ __launch_bounds__(256)
void k_wmma_gemm_abt(const float* __restrict__ A, long sA, int lda,
                     const float* __restrict__ B, long sB, int ldb,
                     void* __restrict__ Cv, long sC, int ldc,
                     int M, int N, int K, int batches) {
  int lane  = threadIdx.x & 31;
  int row15 = lane & 15;
  int hi    = lane >> 4;
  long wid     = ((long)blockIdx.x * blockDim.x + threadIdx.x) >> 5;
  long wstride = ((long)gridDim.x * blockDim.x) >> 5;
  int mt = M >> 4, nt = N >> 4;
  long ntiles = (long)batches * mt * nt;

  for (long t = wid; t < ntiles; t += wstride) {
    int bb  = (int)(t / ((long)mt * nt));
    long r2 = t - (long)bb * mt * nt;
    int im  = (int)(r2 / nt);
    int in_ = (int)(r2 - (long)im * nt);

    const float* Ab = A + (long)bb * sA + (long)(im * 16 + row15) * lda + hi * 8;
    const float* Bb = B + (long)bb * sB + (long)(in_ * 16 + row15) * ldb + hi * 8;

    v8f acc = {};
#pragma unroll 8
    for (int kt = 0; kt < K; kt += 32) {
      const float4* pa = (const float4*)(Ab + kt);
      float4 a0 = pa[0], a1 = pa[1], a2 = pa[4], a3 = pa[5];   // K: kb+0..7, kb+16..23
      const float4* pb = (const float4*)(Bb + kt);
      float4 b0 = pb[0], b1 = pb[1], b2 = pb[4], b3 = pb[5];   // K: kb+0..7, kb+16..23
      v16bf av = pack16(a0, a1, a2, a3);
      v16bf bv = pack16(b0, b1, b2, b3);
      acc = __builtin_amdgcn_wmma_f32_16x16x32_bf16(
          false, av, false, bv, (short)0, acc, false, false);
    }

    int ccol = in_ * 16 + row15;
#pragma unroll
    for (int r = 0; r < 8; ++r) {
      int crow = im * 16 + r + 8 * hi;
      if constexpr (BF16OUT) {
        ((__bf16*)Cv + (long)bb * sC)[(long)crow * ldc + ccol] = (__bf16)acc[r];
      } else {
        ((float*)Cv + (long)bb * sC)[(long)crow * ldc + ccol] = acc[r];
      }
    }
  }
}

// ---------------------------------------------------------------------------
// Persistent scan: one block per batch, 1024 threads (32 waves), 1024 steps.
// ---------------------------------------------------------------------------
__global__ __launch_bounds__(1024, 1)
void k_scan(const float* __restrict__ up, const float* __restrict__ v0,
            const float* __restrict__ Vmat, const float* __restrict__ UpWp,
            const __bf16* __restrict__ WuqT, const __bf16* __restrict__ uqT,
            const __bf16* __restrict__ WvT,  const __bf16* __restrict__ WgT,
            const __bf16* __restrict__ wihT, const __bf16* __restrict__ whhT,
            const float* __restrict__ b_ih,  const float* __restrict__ b_hh,
            float* __restrict__ out) {
  constexpr int NI = 1024, MM = 1024, H = 256, I = 256, G3 = 768;

  __shared__ float ls_d[H], ls_up[I], ls_vv[H], ls_v[H];
  __shared__ float ls_e[MM], ls_c[I], ls_cg[I];
  __shared__ float ls_gi[G3], ls_gh[G3];
  __shared__ float ls_red[32];
  __shared__ float ls_mx, ls_den;

  const int b = blockIdx.x, tid = threadIdx.x, lane = tid & 31, wv = tid >> 5;

  if (tid < H) {
    ls_v[tid]  = v0[b * H + tid];
    ls_vv[tid] = Vmat[b * H + tid];   // V[:,:,0] flat
  }
  __syncthreads();

  const __bf16* Wuqb = WuqT + (long)b * H * MM;
  const __bf16* uqTb = uqT  + (long)b * I * MM;

  for (int i = 0; i < NI; ++i) {
    // ---- phase A: d_h = UpWp[b,i,h] + (v @ Wv^T)_h ; stage up_i row --------
    if (tid < H) {
      float acc = UpWp[((long)b * NI + i) * H + tid];
#pragma unroll 4
      for (int k = 0; k < H; ++k)
        acc = fmaf(ls_v[k], bf2f(WvT[k * H + tid]), acc);   // coalesced in tid
      ls_d[tid]  = acc;
      ls_up[tid] = up[((long)b * NI + i) * I + tid];
    }
    __syncthreads();

    // ---- phase B: s_m = sum_h tanh(WuqT[h][m] + d_h) * Vv_h  (m = tid) ----
    float sm = 0.0f;
    {
      const __bf16* p = Wuqb + tid;
      for (int h0 = 0; h0 < H; h0 += 8) {
        if (h0 + 8 < H) __builtin_prefetch(p + (long)(h0 + 8) * MM, 0, 1);
#pragma unroll
        for (int hh = 0; hh < 8; ++hh) {
          int h = h0 + hh;
          float x = fast_tanh(bf2f(p[(long)h * MM]) + ls_d[h]);
          sm = fmaf(x, ls_vv[h], sm);
        }
      }
    }

    // ---- softmax over m (block-wide max then sum) --------------------------
    float mx = sm;
    for (int o = 16; o > 0; o >>= 1) mx = fmaxf(mx, __shfl_xor(mx, o, 32));
    if (lane == 0) ls_red[wv] = mx;
    __syncthreads();
    if (tid < 32) {
      float m2 = ls_red[tid];
      for (int o = 16; o > 0; o >>= 1) m2 = fmaxf(m2, __shfl_xor(m2, o, 32));
      if (tid == 0) ls_mx = m2;
    }
    __syncthreads();
    float e = __expf(sm - ls_mx);
    ls_e[tid] = e;
    float sum = e;
    for (int o = 16; o > 0; o >>= 1) sum += __shfl_xor(sum, o, 32);
    if (lane == 0) ls_red[wv] = sum;
    __syncthreads();
    if (tid < 32) {
      float s2 = ls_red[tid];
      for (int o = 16; o > 0; o >>= 1) s2 += __shfl_xor(s2, o, 32);
      if (tid == 0) ls_den = s2;
    }
    __syncthreads();
    float rden = 1.0f / ls_den;

    // ---- phase C: c_i = (1/den) * sum_m e_m * uqT[i][m]  (8 i's per wave) --
#pragma unroll
    for (int ii = 0; ii < 8; ++ii) {
      int ci = wv * 8 + ii;
      const __bf16* q = uqTb + (long)ci * MM + lane;
      float part = 0.0f;
#pragma unroll 4
      for (int t = 0; t < 32; ++t)
        part = fmaf(ls_e[lane + 32 * t], bf2f(q[32 * t]), part);
      for (int o = 16; o > 0; o >>= 1) part += __shfl_xor(part, o, 32);
      if (lane == 0) ls_c[ci] = part * rden;
    }
    __syncthreads();

    // ---- phase D: g = sigmoid(r @ Wg^T), r = [up_i, c]; cg = (g*r)[I:] -----
    if (tid < 2 * I) {
      float g = 0.0f;
      for (int k = 0; k < 2 * I; ++k) {
        float rk = (k < I) ? ls_up[k] : ls_c[k - I];
        g = fmaf(rk, bf2f(WgT[k * (2 * I) + tid]), g);      // coalesced in tid
      }
      g = fast_sigmoid(g);
      if (tid >= I) ls_cg[tid - I] = g * ls_c[tid - I];
    }
    __syncthreads();

    // ---- phase E: GRU matvecs gi = cg@w_ih^T + b_ih ; gh = v@w_hh^T + b_hh -
    if (tid < G3) {
      float gi = b_ih[tid], gh = b_hh[tid];
#pragma unroll 2
      for (int k = 0; k < H; ++k) {
        gi = fmaf(ls_cg[k], bf2f(wihT[k * G3 + tid]), gi);
        gh = fmaf(ls_v[k],  bf2f(whhT[k * G3 + tid]), gh);
      }
      ls_gi[tid] = gi;
      ls_gh[tid] = gh;
    }
    __syncthreads();

    // ---- gate combine + state update + output ------------------------------
    if (tid < H) {
      float rg = fast_sigmoid(ls_gi[tid] + ls_gh[tid]);
      float z  = fast_sigmoid(ls_gi[H + tid] + ls_gh[H + tid]);
      float n  = fast_tanh(ls_gi[2 * H + tid] + rg * ls_gh[2 * H + tid]);
      float vn = (1.0f - z) * n + z * ls_v[tid];
      out[((long)i * 64 + b) * H + tid] = vn;                // vs[i][b][h]
      ls_v[tid] = vn;
    }
    __syncthreads();
  }
}

// ---------------------------------------------------------------------------
extern "C" void kernel_launch(void* const* d_in, const int* in_sizes, int n_in,
                              void* d_out, int out_size, void* d_ws, size_t ws_size,
                              hipStream_t stream) {
  (void)in_sizes; (void)n_in; (void)out_size; (void)ws_size;
  const float* up  = (const float*)d_in[0];
  const float* uq  = (const float*)d_in[1];
  const float* v0  = (const float*)d_in[2];
  const float* V   = (const float*)d_in[3];
  const float* Wp  = (const float*)d_in[4];
  const float* Wq  = (const float*)d_in[5];
  const float* Wv  = (const float*)d_in[6];
  const float* Wg  = (const float*)d_in[7];
  const float* wih = (const float*)d_in[8];
  const float* whh = (const float*)d_in[9];
  const float* bih = (const float*)d_in[10];
  const float* bhh = (const float*)d_in[11];

  char* ws = (char*)d_ws;
  size_t off = 0;
  auto alloc = [&](size_t bytes) {
    void* p = ws + off;
    off = (off + bytes + 255) & ~(size_t)255;
    return p;
  };
  float*  UpWp = (float*) alloc(64l * 1024 * 256 * 4);   // 64 MB f32
  __bf16* WuqT = (__bf16*)alloc(64l * 256 * 1024 * 2);   // 32 MB bf16 [b][h][m]
  __bf16* uqT  = (__bf16*)alloc(64l * 256 * 1024 * 2);   // 32 MB bf16 [b][i][m]
  __bf16* WvT  = (__bf16*)alloc(256l * 256 * 2);
  __bf16* WgT  = (__bf16*)alloc(512l * 512 * 2);
  __bf16* wihT = (__bf16*)alloc(256l * 768 * 2);
  __bf16* whhT = (__bf16*)alloc(256l * 768 * 2);

  // Pre-transpose / bf16-convert the streamed operands (one-time).
  k_transpose_f32_bf16<<<2048, 256, 0, stream>>>(uq,  uqT,  64, 1024, 256);
  k_transpose_f32_bf16<<<64,   256, 0, stream>>>(Wv,  WvT,  1,  256,  256);
  k_transpose_f32_bf16<<<128,  256, 0, stream>>>(Wg,  WgT,  1,  512,  512);
  k_transpose_f32_bf16<<<96,   256, 0, stream>>>(wih, wihT, 1,  768,  256);
  k_transpose_f32_bf16<<<96,   256, 0, stream>>>(whh, whhT, 1,  768,  256);

  // WMMA GEMM 1: WuqT[b][h][m] = sum_k Wq[h,k] * uq[b,m,k]   (M=256,N=1024,K=256)
  k_wmma_gemm_abt<true><<<2048, 256, 0, stream>>>(
      Wq, 0l, 256, uq, 1024l * 256, 256,
      (void*)WuqT, 256l * 1024, 1024, 256, 1024, 256, 64);

  // WMMA GEMM 2: UpWp[b][n][h] = sum_k up[b,n,k] * Wp[h,k]   (M=1024,N=256,K=256)
  k_wmma_gemm_abt<false><<<2048, 256, 0, stream>>>(
      up, 1024l * 256, 256, Wp, 0l, 256,
      (void*)UpWp, 1024l * 256, 256, 1024, 256, 256, 64);

  // Persistent recurrent scan: one block per batch sequence.
  k_scan<<<64, 1024, 0, stream>>>(up, v0, V, UpWp, WuqT, uqT, WvT, WgT,
                                  wihT, whhT, bih, bhh, (float*)d_out);
}
